// LSTMTagger_47038481825806
// MI455X (gfx1250) — matrix operand order repaired and checked by reading, not compiled, not run on hardware
//
#include <hip/hip_runtime.h>
#include <hip/hip_bf16.h>
#include <stddef.h>

// ---------------------------------------------------------------------------
// SRU bidirectional tagger on MI455X (gfx1250, wave32, WMMA).
// GEMMs use v_wmma_f32_16x16x32_bf16; everything else is memory-bound
// elementwise/scan work sized for 23.3 TB/s HBM.
// ---------------------------------------------------------------------------

typedef __bf16 bf16_t;
typedef __attribute__((ext_vector_type(16))) __bf16 v16bf;
typedef __attribute__((ext_vector_type(8)))  __bf16 v8bf;
typedef __attribute__((ext_vector_type(8)))  float  v8f;

#define TT 100
#define BB 1000
#define EE 512
#define DD 256
#define ROWS (TT*BB)          // 100000 tokens

// ---------------------------------------------------------------------------
// Weight transpose + f32->bf16: in (batch, K, N) row-major -> out (batch, N, K)
// ---------------------------------------------------------------------------
__global__ __launch_bounds__(256) void transpose_to_bf16(
    const float* __restrict__ in, bf16_t* __restrict__ out, int K, int N)
{
    const size_t total = (size_t)K * N;
    const size_t base  = (size_t)blockIdx.y * total;
    size_t i = (size_t)blockIdx.x * blockDim.x + threadIdx.x;   // out-linear (n*K+k)
    if (i >= total) return;
    int n = (int)(i / K);
    int k = (int)(i % K);
    out[base + i] = (bf16_t)in[base + (size_t)k * N + n];
}

// ---------------------------------------------------------------------------
// Plain f32 -> bf16 convert (head input)
// ---------------------------------------------------------------------------
__global__ __launch_bounds__(256) void f32_to_bf16(
    const float* __restrict__ in, bf16_t* __restrict__ out, size_t n)
{
    size_t i = (size_t)blockIdx.x * blockDim.x + threadIdx.x;
    if (i < n) out[i] = (bf16_t)in[i];
}

// ---------------------------------------------------------------------------
// LayerNorm over E=512, one wave32 per row, writes bf16 A matrix for WMMA.
// ---------------------------------------------------------------------------
__global__ __launch_bounds__(256) void layernorm_to_bf16(
    const float* __restrict__ x, const float* __restrict__ g,
    const float* __restrict__ b, bf16_t* __restrict__ out, int rows)
{
    const int lane = threadIdx.x & 31;
    const int wave = threadIdx.x >> 5;
    const int row  = blockIdx.x * 8 + wave;
    if (row >= rows) return;
    const float* xr = x + (size_t)row * EE;
    float v[16];
    float s = 0.f, ss = 0.f;
#pragma unroll
    for (int i = 0; i < 16; ++i) {
        float t = xr[lane + i * 32];
        v[i] = t; s += t; ss += t * t;
    }
#pragma unroll
    for (int off = 16; off; off >>= 1) {
        s  += __shfl_xor(s,  off, 32);
        ss += __shfl_xor(ss, off, 32);
    }
    const float mean = s * (1.f / EE);
    const float var  = ss * (1.f / EE) - mean * mean;
    const float inv  = rsqrtf(var + 1e-5f);
    bf16_t* orow = out + (size_t)row * EE;
#pragma unroll
    for (int i = 0; i < 16; ++i) {
        int k = lane + i * 32;
        orow[k] = (bf16_t)((v[i] - mean) * inv * g[k] + b[k]);
    }
}

// ---------------------------------------------------------------------------
// WMMA GEMM: C[M,N] = A[M,K] * Bt[N,K]^T  (A bf16 MxK row-major, Bt = weights
// pre-transposed to NxK row-major, C f32). One wave computes a 16(M)x64(N)
// tile: 4 f32 accumulators, v_wmma_f32_16x16x32_bf16 over K.
//
// A fragment (16-bit A 16x32, this code's contiguous K order): lanes 0-15 = rows, element run per lane:
//   lanes 0-15 : K 0..15;  lanes 16-31 : K 16..31
// B fragment (16-bit B 32x16, same contiguous K order as A): lane = column n, contiguous K run:
//   lanes 0-15 : K 0..15;  lanes 16-31 : K 16..31
// ---------------------------------------------------------------------------
__global__ __launch_bounds__(256) void gemm_bf16_wmma(
    const bf16_t* __restrict__ A, const bf16_t* __restrict__ Bt,
    float* __restrict__ C, int M, int N, int K)
{
    const int lane   = threadIdx.x & 31;
    const int wave   = threadIdx.x >> 5;
    const int gw     = blockIdx.x * 8 + wave;
    const int ntiles = N >> 6;                    // 64-wide N tiles
    const int mt     = gw / ntiles;
    const int nt     = gw % ntiles;
    if (mt * 16 >= M) return;
    const int m0 = mt * 16;
    const int n0 = nt * 64;

    const int half = lane >> 4;                   // 0 or 1
    const int lidx = lane & 15;

    const bf16_t* aptr = A  + (size_t)(m0 + lidx) * K + half * 16;
    const bf16_t* bptr = Bt + (size_t)(n0 + lidx) * K + half * 16;

    v8f c0 = {}, c1 = {}, c2 = {}, c3 = {};

    for (int kb = 0; kb < K; kb += 32) {
        v16bf a;
        a = *(const v16bf*)(aptr + kb);                      // K 0..15 / 16..31
        v16bf b0 = *(const v16bf*)(bptr + kb);
        v16bf b1 = *(const v16bf*)(bptr + (size_t)16 * K + kb);
        v16bf b2 = *(const v16bf*)(bptr + (size_t)32 * K + kb);
        v16bf b3 = *(const v16bf*)(bptr + (size_t)48 * K + kb);
        c0 = __builtin_amdgcn_wmma_f32_16x16x32_bf16(false, a, false, b0, (short)0, c0, false, false);
        c1 = __builtin_amdgcn_wmma_f32_16x16x32_bf16(false, a, false, b1, (short)0, c1, false, false);
        c2 = __builtin_amdgcn_wmma_f32_16x16x32_bf16(false, a, false, b2, (short)0, c2, false, false);
        c3 = __builtin_amdgcn_wmma_f32_16x16x32_bf16(false, a, false, b3, (short)0, c3, false, false);
    }

    // C/D layout: lanes 0-15 -> M rows 0..7, lanes 16-31 -> M rows 8..15; n = lane&15
    const int cm = m0 + (lane >> 4) * 8;
    float* cp = C + (size_t)cm * N + n0 + (lane & 15);
#pragma unroll
    for (int r = 0; r < 8; ++r) {
        cp[(size_t)r * N + 0]  = c0[r];
        cp[(size_t)r * N + 16] = c1[r];
        cp[(size_t)r * N + 32] = c2[r];
        cp[(size_t)r * N + 48] = c3[r];
    }
}

// ---------------------------------------------------------------------------
// SRU elementwise recurrence, one thread per (batch b, channel d), serial T.
// U layout: ((t*B + b)*4 + gate)*D + d   (exactly the GEMM's C with N=1024)
// ---------------------------------------------------------------------------
__global__ __launch_bounds__(256) void sru_scan(
    const float* __restrict__ U, const float* __restrict__ v2,
    const float* __restrict__ b2, float* __restrict__ xout, int dir)
{
    const int idx = blockIdx.x * blockDim.x + threadIdx.x;
    if (idx >= BB * DD) return;
    const int b = idx >> 8;
    const int d = idx & 255;
    const float vf  = v2[d];
    const float vr  = v2[DD + d];
    const float bf_ = b2[d];
    const float br  = b2[DD + d];
    float c = 0.f;
    for (int s = 0; s < TT; ++s) {
        const int t = dir ? (TT - 1 - s) : s;
        const float* u = U + ((size_t)t * BB + b) * (4 * DD) + d;
        const float u0 = u[0];
        const float u1 = u[DD];
        const float u2 = u[2 * DD];
        const float u3 = u[3 * DD];
        const float f  = 1.f / (1.f + __expf(-(u1 + vf * c + bf_)));
        c = f * c + (1.f - f) * u0;
        const float r = 1.f / (1.f + __expf(-(u2 + vr * c + br)));
        const float h = r * c + (1.f - r) * u3;
        xout[((size_t)t * BB + b) * EE + dir * DD + d] = h;
    }
}

// ---------------------------------------------------------------------------
// Head epilogue: h -> +b1 -> BN(running stats) -> ReLU -> @W2 + b2 -> logsoftmax
// One wave32 per token row.
// ---------------------------------------------------------------------------
__global__ __launch_bounds__(256) void head_kernel(
    const float* __restrict__ h,  const float* __restrict__ b1,
    const float* __restrict__ bng, const float* __restrict__ bnb,
    const float* __restrict__ bnm, const float* __restrict__ bnv,
    const float* __restrict__ W2,  const float* __restrict__ b2,
    float* __restrict__ out, int rows)
{
    const int lane = threadIdx.x & 31;
    const int wave = threadIdx.x >> 5;
    const int row  = blockIdx.x * 8 + wave;
    if (row >= rows) return;
    const float* hr = h + (size_t)row * EE;
    float a0 = 0.f, a1 = 0.f, a2 = 0.f, a3 = 0.f;
#pragma unroll
    for (int i = 0; i < 16; ++i) {
        const int k = lane + i * 32;
        float t = hr[k] + b1[k];
        t = (t - bnm[k]) * rsqrtf(bnv[k] + 1e-5f) * bng[k] + bnb[k];
        t = fmaxf(t, 0.f);
        a0 += t * W2[k * 4 + 0];
        a1 += t * W2[k * 4 + 1];
        a2 += t * W2[k * 4 + 2];
        a3 += t * W2[k * 4 + 3];
    }
#pragma unroll
    for (int off = 16; off; off >>= 1) {
        a0 += __shfl_xor(a0, off, 32);
        a1 += __shfl_xor(a1, off, 32);
        a2 += __shfl_xor(a2, off, 32);
        a3 += __shfl_xor(a3, off, 32);
    }
    if (lane == 0) {
        const float l0 = a0 + b2[0], l1 = a1 + b2[1];
        const float l2 = a2 + b2[2], l3 = a3 + b2[3];
        const float mx = fmaxf(fmaxf(l0, l1), fmaxf(l2, l3));
        const float se = __expf(l0 - mx) + __expf(l1 - mx) +
                         __expf(l2 - mx) + __expf(l3 - mx);
        const float lse = mx + __logf(se);
        float* o = out + (size_t)row * 4;
        o[0] = l0 - lse; o[1] = l1 - lse; o[2] = l2 - lse; o[3] = l3 - lse;
    }
}

// ---------------------------------------------------------------------------
extern "C" void kernel_launch(void* const* d_in, const int* in_sizes, int n_in,
                              void* d_out, int out_size, void* d_ws, size_t ws_size,
                              hipStream_t stream)
{
    (void)in_sizes; (void)n_in; (void)out_size; (void)ws_size;

    const float* sentence = (const float*)d_in[0];   // (T,B,E)
    const float* sru_W    = (const float*)d_in[1];   // (L,2,E,4D)
    const float* sru_v    = (const float*)d_in[2];   // (L,2,2,D)
    const float* sru_b    = (const float*)d_in[3];   // (L,2,2,D)
    const float* ln_g     = (const float*)d_in[4];   // (L,E)
    const float* ln_b     = (const float*)d_in[5];   // (L,E)
    const float* W1       = (const float*)d_in[6];   // (E,E)
    const float* b1       = (const float*)d_in[7];
    const float* bn_g     = (const float*)d_in[8];
    const float* bn_b     = (const float*)d_in[9];
    const float* bn_mean  = (const float*)d_in[10];
    const float* bn_var   = (const float*)d_in[11];
    const float* W2       = (const float*)d_in[12];  // (E,4)
    const float* b2       = (const float*)d_in[13];
    float* out            = (float*)d_out;           // (T*B, 4)

    // Workspace layout (16/32-byte aligned offsets), ~931 MB total.
    char* ws = (char*)d_ws;
    float*  xbuf0 = (float*)(ws + 0);              // 204.8 MB  (T*B, E)
    float*  xbuf1 = (float*)(ws + 204800000);      // 204.8 MB
    bf16_t* Abf   = (bf16_t*)(ws + 409600000);     // 102.4 MB  (T*B, E) bf16
    float*  U     = (float*)(ws + 512000000);      // 409.6 MB  (T*B, 4D) f32 / head h
    bf16_t* Wt    = (bf16_t*)(ws + 921600000);     //   8.4 MB  sru weights (8, 4D, E) bf16
    bf16_t* W1t   = (bf16_t*)(ws + 930000000);     //   0.5 MB  head W1^T bf16

    // Weight transpose+convert (every call; tiny vs pipeline traffic).
    {
        const int kn = EE * 4 * DD;                // 512*1024
        transpose_to_bf16<<<dim3((kn + 255) / 256, 8), 256, 0, stream>>>(sru_W, Wt, EE, 4 * DD);
        const int kn1 = EE * EE;
        transpose_to_bf16<<<dim3((kn1 + 255) / 256, 1), 256, 0, stream>>>(W1, W1t, EE, EE);
    }

    const float* xin = sentence;
    float* xout = xbuf0;
    for (int l = 0; l < 4; ++l) {
        layernorm_to_bf16<<<ROWS / 8, 256, 0, stream>>>(xin, ln_g + l * EE, ln_b + l * EE, Abf, ROWS);
        for (int dir = 0; dir < 2; ++dir) {
            const bf16_t* Wd = Wt + (size_t)(l * 2 + dir) * (4 * DD) * EE;
            const int waves  = (ROWS / 16) * ((4 * DD) / 64);   // 100000
            gemm_bf16_wmma<<<(waves + 7) / 8, 256, 0, stream>>>(Abf, Wd, U, ROWS, 4 * DD, EE);
            const float* vv = sru_v + (size_t)(l * 2 + dir) * 2 * DD;
            const float* bb = sru_b + (size_t)(l * 2 + dir) * 2 * DD;
            sru_scan<<<(BB * DD) / 256, 256, 0, stream>>>(U, vv, bb, xout, dir);
        }
        xin  = xout;
        xout = (xout == xbuf0) ? xbuf1 : xbuf0;
    }

    // Head: x @ W1 (WMMA) -> fused BN/ReLU/W2/log-softmax.
    f32_to_bf16<<<(ROWS * EE + 255) / 256, 256, 0, stream>>>(xin, Abf, (size_t)ROWS * EE);
    {
        const int waves = (ROWS / 16) * (EE / 64);              // 50000
        gemm_bf16_wmma<<<(waves + 7) / 8, 256, 0, stream>>>(Abf, W1t, U, ROWS, EE, EE);
    }
    head_kernel<<<ROWS / 8, 256, 0, stream>>>(U, b1, bn_g, bn_b, bn_mean, bn_var, W2, b2, out, ROWS);
}
